// SepConv_38663295598621
// MI455X (gfx1250) — matrix operand order repaired and checked by reading, not compiled, not run on hardware
//
#include <hip/hip_runtime.h>

// Problem constants (from reference)
#define B_    2
#define N_    384
#define O_    4
#define C_    128
#define K_    64
#define NK    (N_ * K_)          // 24576 flattened reduction dim
#define BO    (B_ * O_)          // 8 independent GEMMs
#define MB_   3                  // 384 / 128 m-tiles
#define SPLIT 8                  // split-K factor
#define KKPS  (NK / SPLIT)       // 3072 k per split
#define ITERS (KKPS / 32)        // 96 K-steps of 32

#if __has_builtin(__builtin_amdgcn_tensor_load_to_lds)
#define HAVE_TDM 1
#if __has_include(<hip/amd_detail/amd_gfx1250_TDM.h>)
#define TDM_6ARG 1   // amdgpu-toolchain / clang-23 flavor
#endif
#endif

typedef __attribute__((ext_vector_type(16))) __bf16       v16bf;
typedef __attribute__((ext_vector_type(8)))  float        v8f;
typedef __attribute__((ext_vector_type(4)))  float        v4f;
typedef __attribute__((ext_vector_type(4)))  unsigned int v4u;
typedef __attribute__((ext_vector_type(2)))  unsigned int v2u;
typedef __attribute__((ext_vector_type(8)))  int          v8i;
typedef __attribute__((ext_vector_type(4)))  int          v4i;

union Frag { v4u u[2]; v16bf b; };   // 32 bytes = 16 bf16 = 8 VGPRs

// f32x2 -> packed bf16x2 via CDNA5 VOP3P mix ops (RNE), 1 VALU/elem:
//   V_FMA_MIXLO_BF16 / V_FMA_MIXHI_BF16 (ISA 15.10 opcodes 62/63)
__device__ __forceinline__ unsigned int pk2bf(float a, float b) {
  unsigned int d;
  asm("v_fma_mixlo_bf16 %0, %1, 1.0, 0" : "=v"(d) : "v"(a));
  asm("v_fma_mixhi_bf16 %0, %1, 1.0, 0" : "+v"(d) : "v"(b));
  return d;
}

// ---------------------------------------------------------------------------
// rot[p*O+o][c] = sum_k fiber[p,o,k] * W_rot[k,c]   (tiny: 16 blocks)
// ---------------------------------------------------------------------------
__global__ void rot_prep(const float* __restrict__ fiber,
                         const float* __restrict__ Wr,
                         float* __restrict__ rot) {
  int po = blockIdx.x;          // p*O + o
  int c  = threadIdx.x;
  float acc = 0.f;
#pragma unroll
  for (int k = 0; k < K_; ++k)
    acc += fiber[po * K_ + k] * Wr[k * C_ + c];
  rot[po * C_ + c] = acc;
}

// ---------------------------------------------------------------------------
// V (bf16), tiled so each GEMM B-tile (32k x 128c) is one contiguous 8KB blob:
//   Vt[((bo*(NK/32) + kt)*C + c)*32 + k2] = W_spatial[k,c] * x[b,n,o,c]
// ---------------------------------------------------------------------------
__global__ void v_prep(const float* __restrict__ x,
                       const float* __restrict__ Ws,
                       unsigned short* __restrict__ Vt) {
  int blk = blockIdx.x;         // bo*N + n
  int n   = blk % N_;
  int bo  = blk / N_;
  int b = bo >> 2, o = bo & 3;
  int c = threadIdx.x;
  float xv = x[((size_t)(b * N_ + n) * O_ + o) * C_ + c];
#pragma unroll
  for (int j = 0; j < 2; ++j) {  // two 32-k halves of K=64
    unsigned int w[16];
#pragma unroll
    for (int k2 = 0; k2 < 16; ++k2) {
      int k = j * 32 + 2 * k2;
      w[k2] = pk2bf(xv * Ws[k * C_ + c], xv * Ws[(k + 1) * C_ + c]);
    }
    size_t base = ((size_t)(bo * (NK / 32) + n * 2 + j) * C_ + c) * 32;
    v4u* dst = (v4u*)(Vt + base);   // 64B contiguous per thread, 16B aligned
    v4u d0 = {w[0], w[1], w[2], w[3]};    dst[0] = d0;
    v4u d1 = {w[4], w[5], w[6], w[7]};    dst[1] = d1;
    v4u d2 = {w[8], w[9], w[10], w[11]};  dst[2] = d2;
    v4u d3 = {w[12], w[13], w[14], w[15]};dst[3] = d3;
  }
}

// ---------------------------------------------------------------------------
// Split-K WMMA GEMM per (b,o):  Yp[s][bo][m][c] += KB[m, nk] @ V[nk, c]
// Block: 256 thr (8 waves). Output tile 128m x 128c. K-step = 32.
// LDS layouts (padded stride 40 bf16 = 80B per row):
//   ldsA[row][k] : A path = global f32 -> VGPR -> v_fma_mix bf16 -> ds_store
//   ldsB[col][k] : B path = bf16 in ws ->
//        c-rows  0..63 : ONE tensor_load_to_lds (TDM, 1-D tile, HW inserts the
//                        16B/row padding via pad_interval/pad_amount), wave 0
//        c-rows 64..127: per-lane global_load_async_to_lds_b128, waves 4-7
// ---------------------------------------------------------------------------
__global__ __launch_bounds__(256) void gemm_wmma(
    const float* __restrict__ KB,
    const unsigned short* __restrict__ Vt,
    float* __restrict__ Yp) {
  __shared__ __align__(16) unsigned short ldsA[128 * 40];
  __shared__ __align__(16) unsigned short ldsB[128 * 40];

  int bid = blockIdx.x;
  int s   = bid & 7;            // split-K slice
  int mb  = (bid >> 3) % MB_;   // m tile
  int bo  = bid / (SPLIT * MB_);
  int b = bo >> 2, o = bo & 3;

  int t    = threadIdx.x;
  int lane = t & 31, wid = t >> 5;
  int m_off = (wid & 3) * 32;   // wave covers 32 m-rows
  int c_off = (wid >> 2) * 64;  // wave covers 64 c-cols
  int hl = lane >> 4, ln = lane & 15;

  // A-tile global loader: 8 threads per row, float4 each; 4 rows per thread
  int a_q  = t & 7;
  int a_r0 = t >> 3;

  // B-tile async-lane assignment (padded [c][k] LDS layout)
#if HAVE_TDM
  int u    = t & 127;                 // lane job id within waves 4..7
  int cbB  = 64 + (u >> 1);
  int hfB  = u & 1;
  int sOff = 2048 + u * 16;           // element offset into the 4096-elem tile
#else
  int cbB  = t >> 1;
  int hfB  = t & 1;
  int sOff = t * 16;
#endif
  unsigned ldsb_d0 = (unsigned)(uintptr_t)(&ldsB[cbB * 40 + hfB * 16]);
  unsigned ldsb_d1 = ldsb_d0 + 16u;

  v8f acc[2][4];
  v8f zacc = {0.f, 0.f, 0.f, 0.f, 0.f, 0.f, 0.f, 0.f};
#pragma unroll
  for (int mt = 0; mt < 2; ++mt)
#pragma unroll
    for (int ct = 0; ct < 4; ++ct) acc[mt][ct] = zacc;

  const size_t boA = ((size_t)b * N_) * (size_t)N_ * 256;  // b offset into KB

  for (int it = 0; it < ITERS; ++it) {
    int kk = s * KKPS + it * 32;
    int n  = kk >> 6;
    int k0 = kk & 63;

    // ---- stage A globals into registers (overlaps with prior compute) ----
    const float* abase = KB + boA + (size_t)n * 256 + o * 64 + k0 + a_q * 4;
    v4f aR[4];
#pragma unroll
    for (int i = 0; i < 4; ++i) {
      int row = a_r0 + i * 32;
      aR[i] = *(const v4f*)(abase + (size_t)(mb * 128 + row) * 98304);
    }

    // prefetch next K-step's A rows into cache (global_prefetch_b8)
    if (it + 1 < ITERS) {
      int kk2 = kk + 32;
      __builtin_prefetch(KB + boA + (size_t)(kk2 >> 6) * 256 + o * 64 +
                             (kk2 & 63) + a_q * 4 +
                             (size_t)(mb * 128 + a_r0) * 98304,
                         0, 0);
    }

    __syncthreads();  // previous tile fully consumed by all waves

    // ---- B tile: dual DMA paths into LDS ----
    size_t tb = ((size_t)(bo * (NK / 32) + (kk >> 5)) * C_) * 32;  // elements
#if HAVE_TDM
    if (wid == 0) {
      // TDM: 1-D tile of 2048 x 2B elems, pad 4 dwords after every 16 dwords
      unsigned long long ga =
          (unsigned long long)(uintptr_t)Vt + (unsigned long long)tb * 2ull;
      v4u g0 = {1u,                                   // count=1 descriptor
                (unsigned)(uintptr_t)(&ldsB[0]),      // lds_addr
                (unsigned)ga,                         // global_addr[31:0]
                (unsigned)(ga >> 32) | 0x80000000u};  // addr[56:32] | type=2
      v8i g1 = {(int)0x06D10000,  // data_size=2B, pad_en, pad_int=16dw, pad_amt=4dw
                (int)0x08000000,  // tensor_dim0 = 2048 (bits 79:48)
                (int)0x00010000,  // tensor_dim1 = 1
                (int)0x08000000,  // tile_dim0 = 2048 (bits 127:112)
                0,                // tile_dim1=0, tile_dim2=0 (unused)
                2048, 0, 0};      // dim0_stride = 2048 (unused for 1-D)
      v4i gz = {0, 0, 0, 0};
#if TDM_6ARG
      v8i gz8 = {0, 0, 0, 0, 0, 0, 0, 0};
      __builtin_amdgcn_tensor_load_to_lds(g0, g1, gz, gz, gz8, 0);
#else
      __builtin_amdgcn_tensor_load_to_lds(g0, g1, gz, gz, 0);
#endif
    }
    if (wid >= 4)
#endif
    {
      const unsigned short* bsrc = Vt + tb + sOff;
      unsigned long long g0a = (unsigned long long)(uintptr_t)bsrc;
      unsigned long long g1a = g0a + 16ull;
      asm volatile("global_load_async_to_lds_b128 %0, %1, off"
                   :: "v"(ldsb_d0), "v"(g0a) : "memory");
      asm volatile("global_load_async_to_lds_b128 %0, %1, off"
                   :: "v"(ldsb_d1), "v"(g1a) : "memory");
    }

    // ---- A tile: convert f32 -> bf16 (v_fma_mix), store to LDS ----
#pragma unroll
    for (int i = 0; i < 4; ++i) {
      int row = a_r0 + i * 32;
      v2u pk = {pk2bf(aR[i][0], aR[i][1]), pk2bf(aR[i][2], aR[i][3])};
      *(v2u*)(&ldsA[row * 40 + a_q * 4]) = pk;
    }

    // ---- both DMA engines must land before anyone reads ldsB ----
#if HAVE_TDM
    if (wid == 0) __builtin_amdgcn_s_wait_tensorcnt(0);
    if (wid >= 4)
#endif
    {
#if __has_builtin(__builtin_amdgcn_s_wait_asynccnt)
      __builtin_amdgcn_s_wait_asynccnt(0);
#else
      asm volatile("s_wait_asynccnt 0x0" ::: "memory");
#endif
    }
    __syncthreads();

    // ---- WMMA fragment loads (per documented wave32 VGPR layouts) ----
    Frag af[2];
#pragma unroll
    for (int mt = 0; mt < 2; ++mt) {
      int row = m_off + mt * 16 + ln;
      af[mt].u[0] = *(const v4u*)(&ldsA[row * 40 + hl * 8]);
      af[mt].u[1] = *(const v4u*)(&ldsA[row * 40 + 16 + hl * 8]);
    }
#pragma unroll
    for (int ct = 0; ct < 4; ++ct) {
      int col = c_off + ct * 16 + ln;
      Frag bf;
      bf.u[0] = *(const v4u*)(&ldsB[col * 40 + hl * 8]);
      bf.u[1] = *(const v4u*)(&ldsB[col * 40 + 16 + hl * 8]);
#pragma unroll
      for (int mt = 0; mt < 2; ++mt)
        acc[mt][ct] = __builtin_amdgcn_wmma_f32_16x16x32_bf16(
            false, af[mt].b, false, bf.b, (short)0, acc[mt][ct], false, false);
    }
  }

  // ---- write split-K partials (f32) ----
#pragma unroll
  for (int mt = 0; mt < 2; ++mt)
#pragma unroll
    for (int ct = 0; ct < 4; ++ct)
#pragma unroll
      for (int e = 0; e < 8; ++e) {
        int mg = mb * 128 + m_off + mt * 16 + hl * 8 + e;
        int cg = c_off + ct * 16 + ln;
        Yp[(((size_t)s * BO + bo) * N_ + mg) * C_ + cg] = acc[mt][ct][e];
      }
}

// ---------------------------------------------------------------------------
// Fixed-order split-K reduction + fiber (rot) mix + bias.
// ---------------------------------------------------------------------------
__global__ void epilogue(const float* __restrict__ Yp,
                         const float* __restrict__ rot,
                         const float* __restrict__ bias,
                         float* __restrict__ out) {
  int blk = blockIdx.x;         // b*N + m
  int m = blk % N_, b = blk / N_;
  int c = threadIdx.x;
  float yo[O_] = {0.f, 0.f, 0.f, 0.f};
#pragma unroll
  for (int s = 0; s < SPLIT; ++s)
#pragma unroll
    for (int o = 0; o < O_; ++o)
      yo[o] += Yp[(((size_t)s * BO + (b * O_ + o)) * N_ + m) * C_ + c];
  float bc = bias[c];
#pragma unroll
  for (int p = 0; p < O_; ++p) {
    float a = bc;
#pragma unroll
    for (int o = 0; o < O_; ++o)
      a += yo[o] * rot[(p * O_ + o) * C_ + c];
    out[((size_t)(b * N_ + m) * O_ + p) * C_ + c] = a;
  }
}

// ---------------------------------------------------------------------------
extern "C" void kernel_launch(void* const* d_in, const int* in_sizes, int n_in,
                              void* d_out, int out_size, void* d_ws,
                              size_t ws_size, hipStream_t stream) {
  (void)in_sizes; (void)n_in; (void)out_size; (void)ws_size;
  const float* x    = (const float*)d_in[0];
  const float* kb   = (const float*)d_in[1];
  const float* fib  = (const float*)d_in[2];
  const float* Ws   = (const float*)d_in[3];
  const float* Wr   = (const float*)d_in[4];
  const float* bias = (const float*)d_in[5];

  // workspace layout (bytes):
  //   Vt   bf16  BO*NK*C  = 50,331,648
  //   Yp   f32   SPLIT*BO*N*C = 12,582,912
  //   rot  f32   O*O*C    = 8,192          (total ~62.9 MB)
  char* ws = (char*)d_ws;
  unsigned short* Vt = (unsigned short*)ws;
  float* Yp  = (float*)(ws + 50331648u);
  float* rot = (float*)(ws + 50331648u + 12582912u);
  float* out = (float*)d_out;

  rot_prep<<<O_ * O_, C_, 0, stream>>>(fib, Wr, rot);
  v_prep<<<BO * N_, C_, 0, stream>>>(x, Ws, Vt);
  gemm_wmma<<<SPLIT * MB_ * BO, 256, 0, stream>>>(kb, Vt, Yp);
  epilogue<<<B_ * N_, C_, 0, stream>>>(Yp, rot, bias, out);
}
